// ADIAModel_55061480735412
// MI455X (gfx1250) — matrix operand order repaired and checked by reading, not compiled, not run on hardware
//
#include <hip/hip_runtime.h>
#include <hip/hip_fp16.h>

typedef __attribute__((ext_vector_type(16))) _Float16 v16h;
typedef __attribute__((ext_vector_type(8)))  float    v8f;
typedef int v4i __attribute__((vector_size(16)));

#define NLEN 1024
#define XPITCH 68          // halves per transposed row [n][c], 136B stride (conflict-free)
#define HALVES_XT (NLEN * XPITCH)   // 69632
#define WSW_HALVES (3 * 2 * 4 * 32 * 16)  // 12288
#define STAGE_FLOATS (3 * NLEN)           // raw edge_data staging (12KB)

#if __has_builtin(__builtin_amdgcn_global_load_async_to_lds_b128)
#define HAVE_ASYNC_COPY 1
#else
#define HAVE_ASYNC_COPY 0
#endif

__device__ __forceinline__ float gelu_erf(float x) {
    return 0.5f * x * (1.0f + erff(x * 0.70710678118654752f));
}

__device__ __forceinline__ void ln64_stats(const float* a, float& mu, float& rs) {
    float s = 0.f, sq = 0.f;
    #pragma unroll 8
    for (int k = 0; k < 64; ++k) { float v = a[k]; s += v; sq += v * v; }
    mu = s * (1.0f / 64.0f);
    float var = sq * (1.0f / 64.0f) - mu * mu;
    rs = rsqrtf(var + 1e-5f);
}

// ---------------------------------------------------------------------------
// Kernel A: per-edge conv feature extractor. 1 workgroup (8 waves) per edge.
// Whole 64x1024 activation lives in LDS (f16, transposed [n][c]); convs are
// v_wmma_f32_16x16x32_f16 GEMMs (3 taps x 2 K-blocks per 16x16 tile).
// Raw input is staged global->LDS with async-LDS loads (ASYNCcnt) when the
// toolchain exposes the gfx1250 builtin.
// ---------------------------------------------------------------------------
__global__ void __launch_bounds__(256)
conv_extract_kernel(const float* __restrict__ edge_data,
                    const float* __restrict__ stem_w,
                    const float* __restrict__ stem_b,
                    const float* __restrict__ conv_w,
                    const float* __restrict__ gn_g,
                    const float* __restrict__ gn_b,
                    float* __restrict__ emb_out) {
    extern __shared__ char smem[];
    _Float16* xt   = (_Float16*)smem;                 // [1024][68] f16 activation
    _Float16* ht   = xt + HALVES_XT;                  // [1024][68] f16 conv output
    _Float16* wswz = ht + HALVES_XT;                  // swizzled A fragments
    float*    redb = (float*)(wswz + WSW_HALVES);     // [2][4][64] partial stats
    float*    swz  = redb + 512;                      // stem weights+bias (256 f32)
    float*    xstg = swz + 256;                       // raw edge_data staging (3*1024 f32)

    const int tid = threadIdx.x;
    const int m   = blockIdx.x;
    const float* ein = edge_data + (size_t)m * 3 * NLEN;

#if HAVE_ASYNC_COPY
    // ---- async global -> LDS copy of the 12KB raw input (ASYNCcnt path) ----
    {
        char* einm = const_cast<char*>((const char*)ein);
        __attribute__((address_space(1))) v4i* gp =
            (__attribute__((address_space(1))) v4i*)einm;
        __attribute__((address_space(3))) v4i* lp =
            (__attribute__((address_space(3))) v4i*)xstg;
        #pragma unroll
        for (int i = 0; i < 3; ++i) {
            int idx = tid + i * 256;
            __builtin_amdgcn_global_load_async_to_lds_b128(gp + idx, lp + idx, 0, 0);
        }
#if __has_builtin(__builtin_amdgcn_s_wait_asynccnt)
        __builtin_amdgcn_s_wait_asynccnt(0);
#else
        asm volatile("s_wait_asynccnt 0" ::: "memory");
#endif
    }
#endif

    // stage stem weights (3x64) + bias (64)
    if (tid < 192)      swz[tid] = stem_w[tid];
    else                swz[tid] = stem_b[tid - 192];
    __syncthreads();

    // ---- stem: xt[n][d] = sum_c in[c][n] * W[c][d] + b[d] ----
    #pragma unroll
    for (int k = 0; k < 4; ++k) {
        int n = tid + k * 256;
#if HAVE_ASYNC_COPY
        float c0 = xstg[n], c1 = xstg[NLEN + n], c2 = xstg[2 * NLEN + n];
#else
        __builtin_prefetch(ein + ((k + 1) & 3) * 256 + tid, 0, 1);
        float c0 = ein[n], c1 = ein[NLEN + n], c2 = ein[2 * NLEN + n];
#endif
        _Float16* row = xt + n * XPITCH;
        #pragma unroll 8
        for (int d = 0; d < 64; ++d)
            row[d] = (_Float16)(c0 * swz[d] + c1 * swz[64 + d] + c2 * swz[128 + d] + swz[192 + d]);
    }
    __syncthreads();

    const int lane = tid & 31;
    const int wave = tid >> 5;
    const int mo   = wave & 3;        // out-channel tile 0..3
    const int nt0  = wave >> 2;       // col-tile parity
    const int ncol = lane & 15;
    const int kb_b = (lane < 16) ? 0 : 16;   // B-fragment K base
    const int rb_d = mo * 16 + ((lane < 16) ? 0 : 8); // D-fragment row base

    for (int blk = 0; blk < 5; ++blk) {
        // ---- stage swizzled conv weights: wswz[t][kb][mo][lane][16 halves]
        const float* wsrc = conv_w + (size_t)blk * 64 * 64 * 3;
        for (int idx = tid; idx < WSW_HALVES; idx += 256) {
            int h   = idx & 15;
            int l   = (idx >> 4) & 31;
            int moo = (idx >> 9) & 3;
            int kb  = (idx >> 11) & 1;
            int t   = idx >> 12;
            int M   = moo * 16 + (l & 15);
            int K   = kb * 32 + ((l < 16) ? 0 : 16) + h;
            wswz[idx] = (_Float16)wsrc[(M * 64 + K) * 3 + t];
        }
        __syncthreads();

        // ---- hoist the 6 loop-invariant A fragments into registers ----
        v16h afrag[6];
        #pragma unroll
        for (int kb = 0; kb < 2; ++kb)
            #pragma unroll
            for (int t = 0; t < 3; ++t)
                afrag[kb * 3 + t] =
                    *(const v16h*)(wswz + (((t * 2 + kb) * 4 + mo) * 32 + lane) * 16);

        // ---- conv via WMMA: H(64x1024) = sum_t W_t(64x64) * X_shift(t)
        #pragma unroll 2
        for (int nt = nt0; nt < 64; nt += 2) {
            v8f acc = {};
            int n = nt * 16 + ncol;
            #pragma unroll
            for (int kb = 0; kb < 2; ++kb) {
                #pragma unroll
                for (int t = 0; t < 3; ++t) {
                    int sc = n + t - 1;   // 'same' padding
                    union { uint2 u[4]; v16h v; } bfr;
                    if (sc >= 0 && sc < NLEN) {
                        const uint2* bp = (const uint2*)(xt + sc * XPITCH + kb * 32 + kb_b);
                        bfr.u[0] = bp[0]; bfr.u[1] = bp[1]; bfr.u[2] = bp[2]; bfr.u[3] = bp[3];
                    } else {
                        bfr.u[0] = make_uint2(0u, 0u); bfr.u[1] = make_uint2(0u, 0u);
                        bfr.u[2] = make_uint2(0u, 0u); bfr.u[3] = make_uint2(0u, 0u);
                    }
                    acc = __builtin_amdgcn_wmma_f32_16x16x32_f16(
                        false, afrag[kb * 3 + t], false, bfr.v, (short)0, acc, false, false);
                }
            }
            _Float16* hrow = ht + n * XPITCH + rb_d;
            #pragma unroll
            for (int i = 0; i < 8; ++i) hrow[i] = (_Float16)acc[i];
        }
        __syncthreads();

        // ---- GroupNorm stats (deterministic two-stage partial sums)
        {
            int c = tid & 63, q = tid >> 6;
            float s = 0.f, sq = 0.f;
            for (int n = q * 256; n < q * 256 + 256; ++n) {
                float v = (float)ht[n * XPITCH + c];
                s += v; sq += v * v;
            }
            redb[q * 64 + c] = s;
            redb[256 + q * 64 + c] = sq;
        }
        __syncthreads();

        // ---- normalize + GELU + residual into xt
        {
            int c = tid & 63, q = tid >> 6;
            int g = c >> 3;
            float s = 0.f, sq = 0.f;
            #pragma unroll
            for (int j = 0; j < 8; ++j)
                #pragma unroll
                for (int qq = 0; qq < 4; ++qq) {
                    s  += redb[qq * 64 + g * 8 + j];
                    sq += redb[256 + qq * 64 + g * 8 + j];
                }
            float mu  = s * (1.0f / 8192.0f);
            float var = sq * (1.0f / 8192.0f) - mu * mu;
            float rs  = rsqrtf(var + 1e-5f);
            float gg = gn_g[blk * 64 + c], gb = gn_b[blk * 64 + c];
            for (int n = q * 256; n < q * 256 + 256; ++n) {
                float hv = (float)ht[n * XPITCH + c];
                float v  = (hv - mu) * rs * gg + gb;
                float x  = (float)xt[n * XPITCH + c] + gelu_erf(v);
                xt[n * XPITCH + c] = (_Float16)x;
            }
        }
        __syncthreads();
    }

    // ---- mean pool over N -> edge embedding (64 f32)
    {
        int c = tid & 63, q = tid >> 6;
        float s = 0.f;
        for (int n = q * 256; n < q * 256 + 256; ++n) s += (float)xt[n * XPITCH + c];
        redb[q * 64 + c] = s;
    }
    __syncthreads();
    if (tid < 64) {
        float s = redb[tid] + redb[64 + tid] + redb[128 + tid] + redb[192 + tid];
        emb_out[(size_t)m * 64 + tid] = s * (1.0f / 1024.0f);
    }
}

// ---------------------------------------------------------------------------
// Kernel B: type embedding + stat projector + edge merge (per edge, 64 thr)
// ---------------------------------------------------------------------------
__global__ void __launch_bounds__(64)
merge_kernel(const float* __restrict__ emb0, const int* __restrict__ etypes,
             const float* __restrict__ estats, const float* __restrict__ type_tab,
             const float* __restrict__ sp_w1, const float* __restrict__ sp_b1,
             const float* __restrict__ sp_w2, const float* __restrict__ sp_b2,
             const float* __restrict__ sp_ln_g, const float* __restrict__ sp_ln_b,
             const float* __restrict__ em_w, const float* __restrict__ em_b,
             const float* __restrict__ em_ln_g, const float* __restrict__ em_ln_b,
             float* __restrict__ X) {
    int e = blockIdx.x, t = threadIdx.x;
    __shared__ float s1s[64], cat[192], tmp[64];
    float st0 = estats[(size_t)e * 2 + 0], st1 = estats[(size_t)e * 2 + 1];
    float s1 = gelu_erf(st0 * sp_w1[t] + st1 * sp_w1[64 + t] + sp_b1[t]);
    s1s[t] = s1;
    __syncthreads();
    float s2 = sp_b2[t];
    for (int k = 0; k < 64; ++k) s2 += s1s[k] * sp_w2[k * 64 + t];
    tmp[t] = s2;
    __syncthreads();
    float mu, rs; ln64_stats(tmp, mu, rs);
    float stat = (s2 - mu) * rs * sp_ln_g[t] + sp_ln_b[t];
    cat[t]       = emb0[(size_t)e * 64 + t];
    cat[64 + t]  = type_tab[etypes[e] * 64 + t];
    cat[128 + t] = stat;
    __syncthreads();
    float mv = em_b[t];
    for (int k = 0; k < 192; ++k) mv += cat[k] * em_w[k * 64 + t];
    __syncthreads();
    tmp[t] = mv;
    __syncthreads();
    ln64_stats(tmp, mu, rs);
    X[(size_t)e * 64 + t] = gelu_erf((mv - mu) * rs * em_ln_g[t] + em_ln_b[t]);
}

// ---------------------------------------------------------------------------
// Kernel C1: QKV projection (per edge, 64 thr)
// ---------------------------------------------------------------------------
__global__ void __launch_bounds__(64)
qkv_kernel(const float* __restrict__ X, const float* __restrict__ wqkv,
           const float* __restrict__ bqkv, float* __restrict__ qkv) {
    int e = blockIdx.x, t = threadIdx.x;
    __shared__ float xs[64];
    xs[t] = X[(size_t)e * 64 + t];
    __syncthreads();
    for (int j = t; j < 192; j += 64) {
        float s = bqkv[j];
        for (int k = 0; k < 64; ++k) s += xs[k] * wqkv[k * 192 + j];
        qkv[(size_t)e * 192 + j] = s;
    }
}

// ---------------------------------------------------------------------------
// Kernel C2: masked softmax attention, one block per (b,q), 128 thr
// ---------------------------------------------------------------------------
__global__ void __launch_bounds__(128)
attn_kernel(const float* __restrict__ qkv, const unsigned char* __restrict__ mask,
            float* __restrict__ o_out, int B, int E) {
    int b = blockIdx.x / E;
    int q = blockIdx.x % E;
    int t = threadIdx.x;
    __shared__ float sc[384];
    __shared__ float redo[128 * 16];
    __shared__ float sred[128];
    const float* qkvb = qkv + (size_t)b * E * 192;
    for (int h = 0; h < 4; ++h) {
        float qv[16];
        const float* qp = qkvb + (size_t)q * 192 + h * 16;
        #pragma unroll
        for (int i = 0; i < 16; ++i) qv[i] = qp[i];
        for (int k = t; k < E; k += 128) {
            const float* kp = qkvb + (size_t)k * 192 + 64 + h * 16;
            float s = 0.f;
            #pragma unroll
            for (int i = 0; i < 16; ++i) s += qv[i] * kp[i];
            s *= 0.25f;                          // 1/sqrt(16)
            if (!mask[(size_t)b * E + k]) s = -1e30f;
            sc[k] = s;
        }
        __syncthreads();
        float mx = -1e30f;
        for (int k = t; k < E; k += 128) mx = fmaxf(mx, sc[k]);
        sred[t] = mx;
        __syncthreads();
        if (t == 0) {
            float m2 = -1e30f;
            for (int i = 0; i < 128; ++i) m2 = fmaxf(m2, sred[i]);
            sred[0] = m2;
        }
        __syncthreads();
        mx = sred[0];
        __syncthreads();
        float se = 0.f, ov[16];
        #pragma unroll
        for (int i = 0; i < 16; ++i) ov[i] = 0.f;
        for (int k = t; k < E; k += 128) {
            float w = expf(sc[k] - mx);
            const float* vp = qkvb + (size_t)k * 192 + 128 + h * 16;
            se += w;
            #pragma unroll
            for (int i = 0; i < 16; ++i) ov[i] += w * vp[i];
        }
        sred[t] = se;
        #pragma unroll
        for (int i = 0; i < 16; ++i) redo[t * 16 + i] = ov[i];
        __syncthreads();
        if (t == 0) {
            float s2 = 0.f;
            for (int i = 0; i < 128; ++i) s2 += sred[i];
            sred[0] = s2;
        }
        __syncthreads();
        float inv = 1.0f / sred[0];
        if (t < 16) {
            float s = 0.f;
            for (int i = 0; i < 128; ++i) s += redo[i * 16 + t];
            o_out[((size_t)b * E + q) * 64 + h * 16 + t] = s * inv;
        }
        __syncthreads();
    }
}

// ---------------------------------------------------------------------------
// Kernel C3: out-proj + post-norm residual + FFN + post-norm (per edge, 64 thr)
// ---------------------------------------------------------------------------
__global__ void __launch_bounds__(64)
post_kernel(float* __restrict__ X, const float* __restrict__ ob,
            const float* __restrict__ wo, const float* __restrict__ bo,
            const float* __restrict__ g1, const float* __restrict__ b1,
            const float* __restrict__ w1, const float* __restrict__ bb1,
            const float* __restrict__ w2, const float* __restrict__ bb2,
            const float* __restrict__ g2, const float* __restrict__ b2) {
    int e = blockIdx.x, t = threadIdx.x;
    __shared__ float os[64], x1s[64], fs[256], tmp[64];
    os[t]  = ob[(size_t)e * 64 + t];
    tmp[t] = X[(size_t)e * 64 + t];
    __syncthreads();
    float pr = bo[t];
    for (int k = 0; k < 64; ++k) pr += os[k] * wo[k * 64 + t];
    float r = tmp[t] + pr;
    __syncthreads();
    tmp[t] = r;
    __syncthreads();
    float mu, rs; ln64_stats(tmp, mu, rs);
    float x1 = (r - mu) * rs * g1[t] + b1[t];
    x1s[t] = x1;
    __syncthreads();
    for (int j = t; j < 256; j += 64) {
        float s = bb1[j];
        for (int k = 0; k < 64; ++k) s += x1s[k] * w1[k * 256 + j];
        fs[j] = gelu_erf(s);
    }
    __syncthreads();
    float y = bb2[t];
    for (int k = 0; k < 256; ++k) y += fs[k] * w2[k * 64 + t];
    float r2 = x1 + y;
    __syncthreads();
    tmp[t] = r2;
    __syncthreads();
    ln64_stats(tmp, mu, rs);
    X[(size_t)e * 64 + t] = (r2 - mu) * rs * g2[t] + b2[t];
}

// ---------------------------------------------------------------------------
// Kernel D: edge logits head
// ---------------------------------------------------------------------------
__global__ void __launch_bounds__(64)
ehead_kernel(const float* __restrict__ X, const float* __restrict__ w,
             const float* __restrict__ b, float* __restrict__ out) {
    int e = blockIdx.x, t = threadIdx.x;
    __shared__ float xs[64];
    xs[t] = X[(size_t)e * 64 + t];
    __syncthreads();
    if (t < 2) {
        float s = b[t];
        for (int k = 0; k < 64; ++k) s += xs[k] * w[k * 2 + t];
        out[(size_t)e * 2 + t] = s;
    }
}

// ---------------------------------------------------------------------------
// Kernel E: node head (gather 4 directed edges, merge MLP, logits)
// ---------------------------------------------------------------------------
__global__ void __launch_bounds__(64)
nhead_kernel(const float* __restrict__ X, const float* __restrict__ nm_w,
             const float* __restrict__ nm_b, const float* __restrict__ lg,
             const float* __restrict__ lb, const float* __restrict__ nh_w,
             const float* __restrict__ nh_b, float* __restrict__ out,
             int p, int E, int nn) {
    int bi = blockIdx.x / nn, iu = blockIdx.x % nn;
    int u = iu + 2;
    int t = threadIdx.x;
    __shared__ float g[256], nms[64], tmp[64];
    int i0 = u * (p - 1);        // u -> X
    int i1 = u * (p - 1) + 1;    // u -> Y
    int i2 = u - 1;              // X -> u
    int i3 = p + u - 2;          // Y -> u
    size_t base = (size_t)bi * E;
    g[t]        = X[(base + i0) * 64 + t];
    g[64 + t]   = X[(base + i1) * 64 + t];
    g[128 + t]  = X[(base + i2) * 64 + t];
    g[192 + t]  = X[(base + i3) * 64 + t];
    __syncthreads();
    float s = nm_b[t];
    for (int k = 0; k < 256; ++k) s += g[k] * nm_w[k * 64 + t];
    tmp[t] = s;
    __syncthreads();
    float mu, rs; ln64_stats(tmp, mu, rs);
    nms[t] = gelu_erf((s - mu) * rs * lg[t] + lb[t]);
    __syncthreads();
    if (t < 8) {
        float o = nh_b[t];
        for (int k = 0; k < 64; ++k) o += nms[k] * nh_w[k * 8 + t];
        out[((size_t)bi * nn + iu) * 8 + t] = o;
    }
}

// ---------------------------------------------------------------------------
extern "C" void kernel_launch(void* const* d_in, const int* in_sizes, int n_in,
                              void* d_out, int out_size, void* d_ws, size_t ws_size,
                              hipStream_t stream) {
    const float*         edge_data  = (const float*)d_in[0];
    const int*           edge_types = (const int*)d_in[1];
    const unsigned char* edge_mask  = (const unsigned char*)d_in[2];
    const float*         edge_stats = (const float*)d_in[3];
    // d_in[4] is p (device scalar); shapes are recovered from sizes below.
    const float* stem_w  = (const float*)d_in[5];
    const float* stem_b  = (const float*)d_in[6];
    const float* conv_w  = (const float*)d_in[7];
    const float* gn_g    = (const float*)d_in[8];
    const float* gn_b    = (const float*)d_in[9];
    const float* type_tab= (const float*)d_in[10];
    const float* sp_w1   = (const float*)d_in[11];
    const float* sp_b1   = (const float*)d_in[12];
    const float* sp_w2   = (const float*)d_in[13];
    const float* sp_b2   = (const float*)d_in[14];
    const float* sp_ln_g = (const float*)d_in[15];
    const float* sp_ln_b = (const float*)d_in[16];
    const float* em_w    = (const float*)d_in[17];
    const float* em_b    = (const float*)d_in[18];
    const float* em_ln_g = (const float*)d_in[19];
    const float* em_ln_b = (const float*)d_in[20];
    const float* attn_wqkv = (const float*)d_in[21];
    const float* attn_bqkv = (const float*)d_in[22];
    const float* attn_wo   = (const float*)d_in[23];
    const float* attn_bo   = (const float*)d_in[24];
    const float* ln1_g   = (const float*)d_in[25];
    const float* ln1_b   = (const float*)d_in[26];
    const float* ff_w1   = (const float*)d_in[27];
    const float* ff_b1   = (const float*)d_in[28];
    const float* ff_w2   = (const float*)d_in[29];
    const float* ff_b2   = (const float*)d_in[30];
    const float* ln2_g   = (const float*)d_in[31];
    const float* ln2_b   = (const float*)d_in[32];
    const float* eh_w    = (const float*)d_in[33];
    const float* eh_b    = (const float*)d_in[34];
    const float* nm_w    = (const float*)d_in[35];
    const float* nm_b    = (const float*)d_in[36];
    const float* nm_ln_g = (const float*)d_in[37];
    const float* nm_ln_b = (const float*)d_in[38];
    const float* nh_w    = (const float*)d_in[39];
    const float* nh_b    = (const float*)d_in[40];

    // Recover B, p, E from sizes: |edge_types| = B*E with E = p(p-1);
    // node part of d_out = B*(p-2)*8.
    const int BE = in_sizes[1];
    const int node_elems = out_size - BE * 2;
    const int Bnn = node_elems / 8;         // B*(p-2)
    int B = 8, p = 20;
    for (int pp = 3; pp <= 64; ++pp) {
        if (Bnn % (pp - 2)) continue;
        int bb = Bnn / (pp - 2);
        if ((long)bb * pp * (pp - 1) == (long)BE) { B = bb; p = pp; break; }
    }
    const int E = p * (p - 1);
    const int nn = p - 2;

    float* ws   = (float*)d_ws;
    float* emb0 = ws;                       // BE*64
    float* X    = emb0 + (size_t)BE * 64;   // BE*64
    float* qkvb = X    + (size_t)BE * 64;   // BE*192
    float* ob   = qkvb + (size_t)BE * 192;  // BE*64

    const size_t lds_bytes = (size_t)(HALVES_XT * 2 + WSW_HALVES) * 2
                           + (512 + 256 + STAGE_FLOATS) * 4;
    (void)hipFuncSetAttribute(reinterpret_cast<const void*>(conv_extract_kernel),
                              hipFuncAttributeMaxDynamicSharedMemorySize, (int)lds_bytes);

    conv_extract_kernel<<<BE, 256, lds_bytes, stream>>>(
        edge_data, stem_w, stem_b, conv_w, gn_g, gn_b, emb0);

    merge_kernel<<<BE, 64, 0, stream>>>(
        emb0, edge_types, edge_stats, type_tab,
        sp_w1, sp_b1, sp_w2, sp_b2, sp_ln_g, sp_ln_b,
        em_w, em_b, em_ln_g, em_ln_b, X);

    for (int l = 0; l < 2; ++l) {
        qkv_kernel<<<BE, 64, 0, stream>>>(
            X, attn_wqkv + (size_t)l * 64 * 192, attn_bqkv + (size_t)l * 192, qkvb);
        attn_kernel<<<BE, 128, 0, stream>>>(qkvb, edge_mask, ob, B, E);
        post_kernel<<<BE, 64, 0, stream>>>(
            X, ob,
            attn_wo + (size_t)l * 64 * 64, attn_bo + (size_t)l * 64,
            ln1_g + (size_t)l * 64, ln1_b + (size_t)l * 64,
            ff_w1 + (size_t)l * 64 * 256, ff_b1 + (size_t)l * 256,
            ff_w2 + (size_t)l * 256 * 64, ff_b2 + (size_t)l * 64,
            ln2_g + (size_t)l * 64, ln2_b + (size_t)l * 64);
    }

    ehead_kernel<<<BE, 64, 0, stream>>>(X, eh_w, eh_b, (float*)d_out);
    nhead_kernel<<<B * nn, 64, 0, stream>>>(
        X, nm_w, nm_b, nm_ln_g, nm_ln_b, nh_w, nh_b,
        (float*)d_out + (size_t)BE * 2, p, E, nn);
}
